// Attention_2705829396866
// MI455X (gfx1250) — matrix operand order repaired and checked by reading, not compiled, not run on hardware
//
#include <hip/hip_runtime.h>
#include <hip/hip_bf16.h>

// ---------------------------------------------------------------------------
// Types for CDNA5 WMMA (wave32, 16x16x32 f16 -> f32)
// ---------------------------------------------------------------------------
typedef _Float16 f16;
typedef __attribute__((ext_vector_type(16))) _Float16 v16h;
typedef __attribute__((ext_vector_type(8)))  _Float16 v8h;
typedef __attribute__((ext_vector_type(8)))  float    v8f;

#define WMMA_F32_16x16x32_F16(a, b, c) \
    __builtin_amdgcn_wmma_f32_16x16x32_f16(false, (a), false, (b), (short)0, (c), false, false)

// Problem constants (from the reference)
#define BATCH   4
#define CDIM    256
#define NSEQ    2048
#define HEADS   8
#define DHEAD   64
#define HID     512          // HEADS*DHEAD
#define QKVROWS 1536         // 3*HID

// ---------------------------------------------------------------------------
// Tensor Data Mover support (probe-verified builtin; arity differs by clang)
// ---------------------------------------------------------------------------
#if defined(__has_builtin)
#if __has_builtin(__builtin_amdgcn_tensor_load_to_lds) && \
    __has_builtin(__builtin_amdgcn_s_wait_tensorcnt)
#define HAVE_TDM 1
#endif
#endif
#ifndef HAVE_TDM
#define HAVE_TDM 0
#endif

#if HAVE_TDM
typedef unsigned int u32x4 __attribute__((ext_vector_type(4)));
typedef int          i32x4 __attribute__((ext_vector_type(4)));
typedef int          i32x8 __attribute__((ext_vector_type(8)));

// 2D tile DMA: global (row-major, 16-bit elems, row stride in elems) -> LDS,
// rows stored compacted ([tile_h][tile_w] f16).  D# per CDNA5 ISA ch.8.
__device__ __forceinline__ void tdm_load_2d_f16(const void* gaddr,
                                                unsigned lds_off,
                                                unsigned tile_w, unsigned tile_h,
                                                unsigned row_stride,
                                                unsigned tensor_w,
                                                unsigned tensor_h) {
    const unsigned long long ga = (unsigned long long)gaddr;
    u32x4 g0;
    g0[0] = 1u;                                          // count=1, user desc
    g0[1] = lds_off;                                     // lds_addr [63:32]
    g0[2] = (unsigned)(ga & 0xffffffffu);                // global_addr lo
    g0[3] = (unsigned)((ga >> 32) & 0x01ffffffu)         // global_addr[56:32]
          | (2u << 30);                                  // type=2 ("image")
    i32x8 g1;
    g1[0] = (int)(1u << 16);                             // data_size=1 -> 2B
    g1[1] = (int)((tensor_w & 0xffffu) << 16);           // tensor_dim0[15:0]
    g1[2] = (int)((tensor_w >> 16) | ((tensor_h & 0xffffu) << 16));
    g1[3] = (int)((tensor_h >> 16) | (tile_w << 16));    // tile_dim0
    g1[4] = (int)(tile_h & 0xffffu);                     // tile_dim1 (dim2=0)
    g1[5] = (int)row_stride;                             // dim0_stride[31:0]
    g1[6] = 0;                                           // stride hi, dim1_stride
    g1[7] = 0;
    i32x4 z4 = {0, 0, 0, 0};
#if defined(__clang_major__) && (__clang_major__ >= 23)
    i32x8 z8 = {0, 0, 0, 0, 0, 0, 0, 0};
    __builtin_amdgcn_tensor_load_to_lds(g0, g1, z4, z4, z8, 0);
#else
    __builtin_amdgcn_tensor_load_to_lds(g0, g1, z4, z4, 0);
#endif
}
#endif  // HAVE_TDM

// Pack two contiguous v8h (16B LDS reads) into one WMMA half fragment.
__device__ __forceinline__ v16h pack16(v8h lo, v8h hi) {
    v16h a;
#pragma unroll
    for (int t = 0; t < 8; ++t) { a[t] = lo[t]; a[8 + t] = hi[t]; }
    return a;
}

// ---------------------------------------------------------------------------
// f32 -> f16 conversion (weights + activations staged once into workspace)
// ---------------------------------------------------------------------------
__global__ void cvt_f32_to_f16(const float* __restrict__ src,
                               f16* __restrict__ dst, int count) {
    int i = blockIdx.x * blockDim.x + threadIdx.x;
    if (i < count) dst[i] = (f16)src[i];
}

// ---------------------------------------------------------------------------
// A-fragment (16x32 f16) loader from row-major [M,K] memory (K contiguous).
// ---------------------------------------------------------------------------
__device__ __forceinline__ v16h load_a_rowmajor(const f16* __restrict__ A,
                                                int ldA, int mbase, int kbase,
                                                int lane) {
    const f16* ap = A + (size_t)(mbase + (lane & 15)) * ldA + kbase +
                    ((lane & 16) ? 8 : 0);
    return pack16(*(const v8h*)ap, *(const v8h*)(ap + 16));
}

// ---------------------------------------------------------------------------
// WMMA GEMM, f16 output:  C[b] = A x B[b]
//   B staged per 32-K chunk into LDS as [n][k] (transposed) so each wave
//   B-fragment = 2 contiguous ds b128.  Block 256 thr = 8 waves, tile 32x256.
// ---------------------------------------------------------------------------
__global__ __launch_bounds__(256)
void wmma_gemm_f16(const f16* __restrict__ A, const f16* __restrict__ Bm,
                   f16* __restrict__ Cm, int M, int K, int N) {
    __shared__ f16 btile[256 * 32];          // [n(256)][k(32)] = 16 KB
    const int tid = threadIdx.x;
    const int lane = tid & 31;
    const int wave = tid >> 5;
    const int mbase = blockIdx.y * 32 + (wave >> 2) * 16;
    const int nblock = blockIdx.x * 256;
    const int wn = (wave & 3) * 64;
    const f16* Bp = Bm + (size_t)blockIdx.z * K * N;
    f16* Cp = Cm + (size_t)blockIdx.z * M * N;

    const int nc = lane & 15;
    const int koff = (lane & 16) ? 8 : 0;

    v8f acc[4] = {};
    for (int k0 = 0; k0 < K; k0 += 32) {
        __syncthreads();                     // protect btile from prev reads
#pragma unroll
        for (int rep = 0; rep < 4; ++rep) {
            const int idx = tid + rep * 256; // 0..1023
            const int k = idx >> 5;          // 0..31
            const int c8 = (idx & 31) * 8;   // n offset 0..248
            v8h bv = *(const v8h*)(Bp + (size_t)(k0 + k) * N + nblock + c8);
#pragma unroll
            for (int t = 0; t < 8; ++t) btile[(c8 + t) * 32 + k] = bv[t];
        }
        __syncthreads();

        v16h afrag = load_a_rowmajor(A, K, mbase, k0, lane);
#pragma unroll
        for (int t = 0; t < 4; ++t) {
            const f16* bp = btile + (wn + 16 * t + nc) * 32 + koff;
            v16h bfrag = pack16(*(const v8h*)bp, *(const v8h*)(bp + 16));
            acc[t] = WMMA_F32_16x16x32_F16(afrag, bfrag, acc[t]);
        }
    }
    const int rowadd = (lane & 16) ? 8 : 0;
#pragma unroll
    for (int t = 0; t < 4; ++t)
#pragma unroll
        for (int r = 0; r < 8; ++r)
            Cp[(size_t)(mbase + r + rowadd) * N + nblock + wn + 16 * t + nc] =
                (f16)acc[t][r];
}

// Same GEMM but f32 output with bias (final projection straight to d_out).
__global__ __launch_bounds__(256)
void wmma_gemm_f32_bias(const f16* __restrict__ A, const f16* __restrict__ Bm,
                        float* __restrict__ Cm, const float* __restrict__ bias,
                        int M, int K, int N) {
    __shared__ f16 btile[256 * 32];
    const int tid = threadIdx.x;
    const int lane = tid & 31;
    const int wave = tid >> 5;
    const int mbase = blockIdx.y * 32 + (wave >> 2) * 16;
    const int nblock = blockIdx.x * 256;
    const int wn = (wave & 3) * 64;
    const f16* Bp = Bm + (size_t)blockIdx.z * K * N;
    float* Cp = Cm + (size_t)blockIdx.z * M * N;

    const int nc = lane & 15;
    const int koff = (lane & 16) ? 8 : 0;

    v8f acc[4] = {};
    for (int k0 = 0; k0 < K; k0 += 32) {
        __syncthreads();
#pragma unroll
        for (int rep = 0; rep < 4; ++rep) {
            const int idx = tid + rep * 256;
            const int k = idx >> 5;
            const int c8 = (idx & 31) * 8;
            v8h bv = *(const v8h*)(Bp + (size_t)(k0 + k) * N + nblock + c8);
#pragma unroll
            for (int t = 0; t < 8; ++t) btile[(c8 + t) * 32 + k] = bv[t];
        }
        __syncthreads();

        v16h afrag = load_a_rowmajor(A, K, mbase, k0, lane);
#pragma unroll
        for (int t = 0; t < 4; ++t) {
            const f16* bp = btile + (wn + 16 * t + nc) * 32 + koff;
            v16h bfrag = pack16(*(const v8h*)bp, *(const v8h*)(bp + 16));
            acc[t] = WMMA_F32_16x16x32_F16(afrag, bfrag, acc[t]);
        }
    }
    const int rowadd = (lane & 16) ? 8 : 0;
#pragma unroll
    for (int t = 0; t < 4; ++t)
#pragma unroll
        for (int r = 0; r < 8; ++r) {
            const int row = mbase + r + rowadd;
            Cp[(size_t)row * N + nblock + wn + 16 * t + nc] =
                acc[t][r] + bias[row];
        }
}

// ---------------------------------------------------------------------------
// Fused flash attention over qkv f16 buffer [b, 1536, n] (channel-major).
// Block = 128 threads = 4 waves; block owns 64 queries, streams K/V in
// 32-key chunks with online softmax (scores never touch memory).
// V tile is DMA'd global->LDS by the Tensor Data Mover (wave 0 issues the
// descriptor, s_wait_tensorcnt completes it) overlapping the VALU-side K
// transpose; fragments are all contiguous ds_load_b128 pairs.
// ---------------------------------------------------------------------------
__global__ __launch_bounds__(128)
void flash_attn(const f16* __restrict__ qkv, f16* __restrict__ outb, int n) {
    __shared__ f16 qtile[64 * 64];       // [query][d]   8 KB (Q^T, pre-scaled)
    __shared__ f16 ktile[32 * 64];       // [key][d]     4 KB (K^T)
    __shared__ f16 vtile[64 * 32];       // [d][key]     4 KB (natural)
    __shared__ f16 pbuf[4][16 * 32];     // per-wave P transpose staging, 4 KB

    const int tid = threadIdx.x;
    const int lane = tid & 31;
    const int wave = tid >> 5;
    const int bb = blockIdx.z;
    const int h = blockIdx.y;
    const int qblock = blockIdx.x * 64;

    const f16* qp = qkv + ((size_t)bb * QKVROWS + h * DHEAD) * n;
    const f16* kp = qkv + ((size_t)bb * QKVROWS + HID + h * DHEAD) * n;
    const f16* vp = qkv + ((size_t)bb * QKVROWS + 2 * HID + h * DHEAD) * n;

    const int nc = lane & 15;
    const int koff = (lane & 16) ? 8 : 0;
    const int rowadd = (lane & 16) ? 8 : 0;
    const _Float16 sc = (_Float16)0.125f;        // DIM_HEAD^-0.5

    // ---- stage Q^T once: [64 q][64 d], scaled; coalesced b128 loads ----
#pragma unroll
    for (int rep = 0; rep < 4; ++rep) {
        const int idx = tid + rep * 128;         // 0..511
        const int d = idx >> 3;                  // 0..63
        const int c8 = (idx & 7) * 8;            // query offset 0..56
        v8h qv = *(const v8h*)(qp + (size_t)d * n + qblock + c8);
        qv = qv * sc;
#pragma unroll
        for (int t = 0; t < 8; ++t) qtile[(c8 + t) * 64 + d] = qv[t];
    }
    __syncthreads();

    // ---- load Q A-fragments from LDS (contiguous b128 pairs) ----
    v16h qf[2];
#pragma unroll
    for (int c = 0; c < 2; ++c) {
        const f16* ap = qtile + (wave * 16 + nc) * 64 + c * 32 +
                        ((lane & 16) ? 8 : 0);
        qf[c] = pack16(*(const v8h*)ap, *(const v8h*)(ap + 16));
    }

    float mrow[8], lrow[8];
    v8f oacc[4] = {};
#pragma unroll
    for (int r = 0; r < 8; ++r) { mrow[r] = -3.0e38f; lrow[r] = 0.0f; }

    for (int jb = 0; jb < n; jb += 32) {
        if (jb + 32 < n) {
            __builtin_prefetch(kp + jb + 32, 0, 1);      // global_prefetch_b8
            __builtin_prefetch(vp + jb + 32, 0, 1);
        }
        __syncthreads();                 // tiles free (prev iter reads done)

#if HAVE_TDM
        // V tile [64 d][32 key] via async Tensor Data Mover (no transpose
        // needed: memory layout already matches O-WMMA fragment direction)
        if (wave == 0)
            tdm_load_2d_f16(vp + jb, (unsigned)(unsigned long long)(uintptr_t)vtile,
                            /*tile_w=*/32, /*tile_h=*/64,
                            /*row_stride=*/(unsigned)n,
                            /*tensor_w=*/(unsigned)n, /*tensor_h=*/64);
#endif
        // ---- stage K^T [32 key][64 d] (VALU transpose, overlaps TDM) ----
#pragma unroll
        for (int rep = 0; rep < 2; ++rep) {
            const int idx = tid + rep * 128;             // 0..255
            const int d = idx >> 2;                      // 0..63
            const int c8 = (idx & 3) * 8;                // key offset 0..24
            v8h kv = *(const v8h*)(kp + (size_t)d * n + jb + c8);
#pragma unroll
            for (int t = 0; t < 8; ++t) ktile[(c8 + t) * 64 + d] = kv[t];
#if !HAVE_TDM
            v8h vv = *(const v8h*)(vp + (size_t)d * n + jb + c8);
            *(v8h*)(vtile + d * 32 + c8) = vv;           // natural, b128
#endif
        }
#if HAVE_TDM
        if (wave == 0) __builtin_amdgcn_s_wait_tensorcnt(0);
#endif
        __syncthreads();

        // ---- scores S (16 q x 32 keys), f32 accumulate, K-dim d=64 ----
        v8f s0 = {}, s1 = {};
#pragma unroll
        for (int c = 0; c < 2; ++c) {
            {   // keys jb..jb+15
                const f16* bp = ktile + nc * 64 + c * 32 + koff;
                v16h kf = pack16(*(const v8h*)bp, *(const v8h*)(bp + 16));
                s0 = WMMA_F32_16x16x32_F16(qf[c], kf, s0);
            }
            {   // keys jb+16..jb+31
                const f16* bp = ktile + (16 + nc) * 64 + c * 32 + koff;
                v16h kf = pack16(*(const v8h*)bp, *(const v8h*)(bp + 16));
                s1 = WMMA_F32_16x16x32_F16(qf[c], kf, s1);
            }
        }

        // ---- online softmax (row = C-layout: VGPR idx r + lane half) ----
        float alpha[8];
#pragma unroll
        for (int r = 0; r < 8; ++r) {
            float mx = fmaxf(s0[r], s1[r]);
#pragma unroll
            for (int off = 1; off < 16; off <<= 1)
                mx = fmaxf(mx, __shfl_xor(mx, off, 32));
            const float mnew = fmaxf(mrow[r], mx);
            alpha[r] = __expf(mrow[r] - mnew);
            mrow[r] = mnew;
            const float p0 = __expf(s0[r] - mnew);
            const float p1 = __expf(s1[r] - mnew);
            float rs = p0 + p1;
#pragma unroll
            for (int off = 1; off < 16; off <<= 1)
                rs += __shfl_xor(rs, off, 32);
            lrow[r] = lrow[r] * alpha[r] + rs;
            s0[r] = p0;
            s1[r] = p1;
        }

        // ---- P: C-layout -> A-layout through per-wave LDS tile ----
        f16* pb = pbuf[wave];
#pragma unroll
        for (int r = 0; r < 8; ++r) {
            pb[(r + rowadd) * 32 + nc]      = (f16)s0[r];
            pb[(r + rowadd) * 32 + 16 + nc] = (f16)s1[r];
        }
        asm volatile("s_wait_dscnt 0" ::: "memory");   // per-wave LDS fence
        v16h pf;
        {
            const f16* pap = pb + nc * 32 + ((lane & 16) ? 8 : 0);
            pf = pack16(*(const v8h*)pap, *(const v8h*)(pap + 16));
        }

        // rescale running O, then accumulate P(16x32) x V^T(32x64)
#pragma unroll
        for (int c = 0; c < 4; ++c)
#pragma unroll
            for (int r = 0; r < 8; ++r) oacc[c][r] *= alpha[r];

#pragma unroll
        for (int c = 0; c < 4; ++c) {
            const f16* vb = vtile + (c * 16 + nc) * 32 + koff;
            v16h vf = pack16(*(const v8h*)vb, *(const v8h*)(vb + 16));
            oacc[c] = WMMA_F32_16x16x32_F16(pf, vf, oacc[c]);
        }
    }

    // ---- epilogue: normalize, write back channel-major [d, n] ----
    f16* op = outb + ((size_t)bb * HID + h * DHEAD) * n;
#pragma unroll
    for (int c = 0; c < 4; ++c)
#pragma unroll
        for (int r = 0; r < 8; ++r) {
            const float val = oacc[c][r] / lrow[r];
            op[(size_t)(c * 16 + nc) * n + (qblock + wave * 16 + r + rowadd)] =
                (f16)val;
        }
}

// ---------------------------------------------------------------------------
// Launch
// ---------------------------------------------------------------------------
extern "C" void kernel_launch(void* const* d_in, const int* in_sizes, int n_in,
                              void* d_out, int out_size, void* d_ws, size_t ws_size,
                              hipStream_t stream) {
    const float* x_f32    = (const float*)d_in[0];   // [4,256,2048]
    const float* wqkv_f32 = (const float*)d_in[1];   // [1536,256]
    const float* wout_f32 = (const float*)d_in[2];   // [256,512]
    const float* bout_f32 = (const float*)d_in[3];   // [256]
    float* out = (float*)d_out;                       // [4,256,2048]

    // Workspace carve-up (all f16, 16B-aligned sizes)
    char* ws = (char*)d_ws;
    const size_t cnt_x    = (size_t)BATCH * CDIM * NSEQ;      // 2,097,152
    const size_t cnt_wqkv = (size_t)QKVROWS * CDIM;           //   393,216
    const size_t cnt_wout = (size_t)CDIM * HID;               //   131,072
    const size_t cnt_qkv  = (size_t)BATCH * QKVROWS * NSEQ;   // 12,582,912
    const size_t cnt_ao   = (size_t)BATCH * HID * NSEQ;       //  4,194,304

    f16* x_h    = (f16*)ws;                    ws += cnt_x * sizeof(f16);
    f16* wqkv_h = (f16*)ws;                    ws += cnt_wqkv * sizeof(f16);
    f16* wout_h = (f16*)ws;                    ws += cnt_wout * sizeof(f16);
    f16* qkv_h  = (f16*)ws;                    ws += cnt_qkv * sizeof(f16);
    f16* ao_h   = (f16*)ws;                    ws += cnt_ao * sizeof(f16);

    // 1) stage f16 copies
    cvt_f32_to_f16<<<(int)((cnt_x + 255) / 256), 256, 0, stream>>>(x_f32, x_h, (int)cnt_x);
    cvt_f32_to_f16<<<(int)((cnt_wqkv + 255) / 256), 256, 0, stream>>>(wqkv_f32, wqkv_h, (int)cnt_wqkv);
    cvt_f32_to_f16<<<(int)((cnt_wout + 255) / 256), 256, 0, stream>>>(wout_f32, wout_h, (int)cnt_wout);

    // 2) QKV projection: [1536,256] x [256,2048] per batch
    {
        dim3 grid(NSEQ / 256, QKVROWS / 32, BATCH);
        wmma_gemm_f16<<<grid, 256, 0, stream>>>(wqkv_h, x_h, qkv_h,
                                                QKVROWS, CDIM, NSEQ);
    }

    // 3) fused flash attention
    {
        dim3 grid(NSEQ / 64, HEADS, BATCH);
        flash_attn<<<grid, 128, 0, stream>>>(qkv_h, ao_h, NSEQ);
    }

    // 4) output projection + bias, f32 straight into d_out
    {
        dim3 grid(NSEQ / 256, CDIM / 32, BATCH);
        wmma_gemm_f32_bias<<<grid, 256, 0, stream>>>(wout_h, ao_h, out,
                                                     bout_f32, CDIM, HID, NSEQ);
    }
}
